// DGCNNSeg_53532472377822
// MI455X (gfx1250) — matrix operand order repaired and checked by reading, not compiled, not run on hardware
//
#include <hip/hip_runtime.h>
#include <hip/hip_bf16.h>
#include <cstdint>
#include <cstddef>

// ---------------------------------------------------------------------------
// DGCNN segmentation forward for MI455X (gfx1250, wave32, WMMA + TDM).
// All GEMM-like work (KNN Gram, edge MLPs, conv5, head) runs on
// v_wmma_f32_16x16x32_f16.  BN folded into weights on device.  GEMM weight
// tiles are staged to LDS with the Tensor Data Mover (tensor_load_to_lds,
// s_wait_tensorcnt), double-buffered per wave.
// ---------------------------------------------------------------------------

typedef __attribute__((ext_vector_type(16))) _Float16     v16h;
typedef __attribute__((ext_vector_type(8)))  _Float16     v8h;
typedef __attribute__((ext_vector_type(8)))  float        v8f;
typedef __attribute__((ext_vector_type(4)))  unsigned int v4u;
typedef __attribute__((ext_vector_type(4)))  int          v4i;
typedef __attribute__((ext_vector_type(8)))  int          v8i;

#define DEV __device__ __forceinline__

static constexpr int BB   = 2;
static constexpr int NN   = 4096;
static constexpr int KNBR = 20;

#if __has_builtin(__builtin_amdgcn_tensor_load_to_lds)
#define HAVE_TDM 1
#endif

DEV v8h ld8h(const _Float16* p) { return *reinterpret_cast<const v8h*>(p); }

DEV v16h make16(v8h a, v8h b) {
  v16h r;
#pragma unroll
  for (int i = 0; i < 8; ++i) { r[i] = a[i]; r[i + 8] = b[i]; }
  return r;
}

// A-fragment (16x32 f16).  Layout: lane<16 -> row=lane, K {0..15};
// lane>=16 -> row=lane-16, K {16..31}.  base points at (row0, k0).
// ldh==0 broadcasts a single row (used for the global-feature concat).
DEV v16h load_a(const _Float16* base, int ldh) {
  const int lane = threadIdx.x & 31;
  const int row  = lane & 15;
  const int koff = (lane < 16) ? 0 : 16;
  const _Float16* p = base + (size_t)row * ldh + koff;
  return make16(ld8h(p), ld8h(p + 8));
}

// B-fragment (32x16 f16) from row-major weights: column n of B = row n of W,
// per-lane data is 16 CONTIGUOUS halves.
DEV v16h load_b(const _Float16* W, int kpad, int n0, int k0) {
  const int lane = threadIdx.x & 31;
  const int row  = n0 + (lane & 15);
  const int koff = k0 + ((lane < 16) ? 0 : 16);
  const _Float16* p = W + (size_t)row * kpad + koff;
  return make16(ld8h(p), ld8h(p + 8));
}

// B-fragment from an LDS-staged (rows x 32) tile.
DEV v16h load_b_lds(const _Float16* tile, int j) {
  const int lane = threadIdx.x & 31;
  const int row  = j * 16 + (lane & 15);
  const int koff = (lane < 16) ? 0 : 16;
  const _Float16* p = tile + row * 32 + koff;
  return make16(ld8h(p), ld8h(p + 8));
}

DEV v8f wmma(v16h a, v16h b, v8f c) {
  return __builtin_amdgcn_wmma_f32_16x16x32_f16(false, a, false, b, (short)0, c,
                                                false, false);
}

DEV float lrelu(float v) { return (v >= 0.f) ? v : 0.2f * v; }

// ---------------------------------------------------------------------------
// Tensor Data Mover: issue a 2D tile load (rows x rowHalves f16, row stride
// strideHalves) from global into LDS at ldsOff.  D# layout per CDNA5 ISA
// 8.3/8.4: group0 = {count, lds_addr, global_addr, type=2}; group1 carries
// data_size=2B, tensor dims (== tile dims, no OOB), tile dims, dim0 stride.
// ---------------------------------------------------------------------------
DEV void tdm_load_tile(unsigned ldsOff, const _Float16* gsrc, int rowHalves,
                       int rows, long strideHalves) {
#ifdef HAVE_TDM
  const unsigned long long ga = (unsigned long long)(uintptr_t)gsrc;
  v4u g0;
  g0.x = 1u;                                            // count=1 (user D#)
  g0.y = ldsOff;                                        // lds_addr (bytes)
  g0.z = (unsigned)ga;                                  // global_addr[31:0]
  g0.w = (unsigned)((ga >> 32) & 0x01ffffffu) | (2u << 30);  // addr[56:32]|type=2
  const unsigned d0 = (unsigned)rowHalves;
  const unsigned d1 = (unsigned)rows;
  const unsigned long long st = (unsigned long long)strideHalves;
  v8i g1;
  g1[0] = (int)(1u << 16);                              // data_size = 1 -> 2B
  g1[1] = (int)((d0 & 0xffffu) << 16);                  // tensor_dim0 lo
  g1[2] = (int)((d0 >> 16) | ((d1 & 0xffffu) << 16));   // dim0 hi | dim1 lo
  g1[3] = (int)((d1 >> 16) | (d0 << 16));               // dim1 hi | tile_dim0
  g1[4] = (int)(d1 & 0xffffu);                          // tile_dim1 (tile_dim2=0)
  g1[5] = (int)(unsigned)(st & 0xffffffffu);            // dim0_stride lo
  g1[6] = (int)(unsigned)((st >> 32) & 0xffffu);        // dim0_stride hi
  g1[7] = 0;
  const v4i z = {0, 0, 0, 0};
#if __clang_major__ >= 23
  const v8i z8 = {0, 0, 0, 0, 0, 0, 0, 0};
  __builtin_amdgcn_tensor_load_to_lds(g0, g1, z, z, z8, 0);
#else
  __builtin_amdgcn_tensor_load_to_lds(g0, g1, z, z, 0);
#endif
#else
  (void)ldsOff; (void)gsrc; (void)rowHalves; (void)rows; (void)strideHalves;
#endif
}

// ---------------------------------------------------------------------------
// Weight prep: fold BN (eval) into W and bias, convert to f16, pad.
// ---------------------------------------------------------------------------
__global__ void pack_weight(const float* __restrict__ W, int cout, int cin,
                            const float* __restrict__ g, const float* __restrict__ bb,
                            const float* __restrict__ m, const float* __restrict__ v,
                            const float* __restrict__ bplain,
                            _Float16* __restrict__ Wh, int kpad, int coutpad,
                            float* __restrict__ bias) {
  const int i = blockIdx.x * 256 + threadIdx.x;
  if (i >= coutpad * kpad) return;
  const int o = i / kpad, c = i - o * kpad;
  float s = 1.f, bi = 0.f;
  if (o < cout) {
    if (g) { const float inv = rsqrtf(v[o] + 1e-5f); s = g[o] * inv; bi = bb[o] - m[o] * s; }
    else if (bplain) { bi = bplain[o]; }
  }
  const float val = (o < cout && c < cin) ? W[(size_t)o * cin + c] * s : 0.f;
  Wh[i] = (_Float16)val;
  if (c == 0) bias[o] = bi;
}

// (B,3,N) f32 -> (B*N,32) f16 channel-padded + squared norms.
__global__ void prep_x(const float* __restrict__ x, _Float16* __restrict__ xh,
                       float* __restrict__ sq) {
  const int i = blockIdx.x * 256 + threadIdx.x;
  if (i >= BB * NN) return;
  const int b = i / NN, n = i - b * NN;
  float s = 0.f;
#pragma unroll
  for (int c = 0; c < 32; ++c) {
    const float v = (c < 3) ? x[(size_t)b * 3 * NN + (size_t)c * NN + n] : 0.f;
    if (c < 3) s += v * v;
    xh[(size_t)i * 32 + c] = (_Float16)v;
  }
  sq[i] = s;
}

__global__ void compute_sq(const _Float16* __restrict__ A, int ld, int C,
                           float* __restrict__ sq) {
  const int i = blockIdx.x * 256 + threadIdx.x;
  if (i >= BB * NN) return;
  const _Float16* p = A + (size_t)i * ld;
  float s = 0.f;
  for (int c = 0; c < C; ++c) { const float v = (float)p[c]; s += v * v; }
  sq[i] = s;
}

// Global max-pool over N, split for bandwidth: 16 partial chunks then reduce.
__global__ void colmax_part(const _Float16* __restrict__ lf, _Float16* __restrict__ pm) {
  const int t = blockIdx.x * 256 + threadIdx.x;
  if (t >= BB * 16 * 1024) return;
  const int b = t / (16 * 1024);
  const int rest = t - b * 16 * 1024;
  const int chunk = rest / 1024, c = rest - chunk * 1024;
  float m = -3.4e38f;
  const int n0 = chunk * (NN / 16);
  for (int n = n0; n < n0 + NN / 16; ++n)
    m = fmaxf(m, (float)lf[((size_t)b * NN + n) * 1024 + c]);
  pm[t] = (_Float16)m;
}

__global__ void colmax_fin(const _Float16* __restrict__ pm, _Float16* __restrict__ gf) {
  const int t = blockIdx.x * 256 + threadIdx.x;
  if (t >= BB * 1024) return;
  const int b = t / 1024, c = t - b * 1024;
  float m = -3.4e38f;
#pragma unroll
  for (int k = 0; k < 16; ++k)
    m = fmaxf(m, (float)pm[(size_t)b * 16 * 1024 + k * 1024 + c]);
  gf[t] = (_Float16)m;
}

// ---------------------------------------------------------------------------
// KNN: one wave owns 32 query rows.  Query A-fragments are loop-invariant
// (live in VGPRs); candidate B-fragments stream all N points.  16x16 Gram
// tile via WMMA -> LDS -> per-lane guarded top-20 insertion (register list).
// pd = 2*x.y - |x|^2 - |y|^2; self gives 0 == max, matching jax top_k.
// ---------------------------------------------------------------------------
template <int CPAD>
__global__ __launch_bounds__(128) void knn_kernel(const _Float16* __restrict__ X,
                                                  int ldh,
                                                  const float* __restrict__ sq,
                                                  int* __restrict__ idxOut) {
  constexpr int KT = CPAD / 32;
  __shared__ float tile[4][512];
  const int wid  = (blockIdx.x << 2) + (threadIdx.x >> 5);
  const int lane = threadIdx.x & 31;
  const int rowg = wid * 32;          // global row over B*N
  const int b    = rowg / NN;
  const int n0   = rowg - b * NN;
  const _Float16* Xb = X + (size_t)b * NN * ldh;
  const float*    Sb = sq + (size_t)b * NN;

  v16h a0[KT], a1[KT];
#pragma unroll
  for (int kt = 0; kt < KT; ++kt) {
    a0[kt] = load_a(Xb + (size_t)n0 * ldh + kt * 32, ldh);
    a1[kt] = load_a(Xb + (size_t)(n0 + 16) * ldh + kt * 32, ldh);
  }
  const float sqr = Sb[n0 + lane];

  float bv[KNBR]; int bidx[KNBR];
#pragma unroll
  for (int t = 0; t < KNBR; ++t) { bv[t] = -3.4e38f; bidx[t] = 0; }

  float* my   = tile[threadIdx.x >> 5];
  const int c  = lane & 15;
  const int mb = (lane < 16) ? 0 : 8;

  for (int ct = 0; ct < NN / 16; ++ct) {
    const int colbase = ct * 16;
    __builtin_prefetch(Xb + (size_t)(colbase + 32) * ldh, 0, 0);
    v8f acc0 = {}; v8f acc1 = {};
#pragma unroll
    for (int kt = 0; kt < KT; ++kt) {
      const v16h bf = load_b(Xb, ldh, colbase, kt * 32);
      acc0 = wmma(a0[kt], bf, acc0);
      acc1 = wmma(a1[kt], bf, acc1);
    }
#pragma unroll
    for (int i = 0; i < 8; ++i) {
      my[(mb + i) * 16 + c]      = acc0[i];
      my[(16 + mb + i) * 16 + c] = acc1[i];
    }
    __syncthreads();
    const int r = lane;
#pragma unroll
    for (int j = 0; j < 16; ++j) {
      const float g  = my[r * 16 + j];
      const float pd = 2.f * g - sqr - Sb[colbase + j];
      if (pd > bv[KNBR - 1]) {
        float cv = pd; int ci = colbase + j;
#pragma unroll
        for (int t = 0; t < KNBR; ++t) {
          if (cv > bv[t]) {
            const float tv = bv[t]; bv[t] = cv; cv = tv;
            const int   ti = bidx[t]; bidx[t] = ci; ci = ti;
          }
        }
      }
    }
    __syncthreads();
  }
  int* op = idxOut + (size_t)(rowg + lane) * KNBR;
#pragma unroll
  for (int t = 0; t < KNBR; ++t) op[t] = bidx[t];
}

// ---------------------------------------------------------------------------
// EdgeConv: one wave per point.  Build e=[nbr-x, x] (K padded 20->32 by
// duplicating edge 19 -> max-neutral) in LDS, two WMMA MLP stages through
// LDS, masked max over K via in-lane reduce + shfl_xor(16).
// ---------------------------------------------------------------------------
template <int CIN, int CPAD2, int COUT>
__global__ __launch_bounds__(64) void edge_kernel(
    const _Float16* __restrict__ X, int ldin, const int* __restrict__ knn,
    const _Float16* __restrict__ W1, const float* __restrict__ b1,
    const _Float16* __restrict__ W2, const float* __restrict__ b2,
    _Float16* __restrict__ out, int ldout) {
  __shared__ alignas(16) _Float16 eT[2][32 * CPAD2];
  __shared__ alignas(16) _Float16 hT[2][32 * COUT];
  const int w    = threadIdx.x >> 5;
  const int lane = threadIdx.x & 31;
  const int p    = (blockIdx.x << 1) + w;
  const int b    = p / NN;
  const int n    = p - b * NN;

  // stage 0: gather edge features (each lane builds one of 32 edge rows)
  const int r   = lane;
  const int rr  = (r < KNBR) ? r : (KNBR - 1);
  const int nbr = knn[(size_t)p * KNBR + rr];
  const _Float16* xrow = X + (size_t)(b * NN + n) * ldin;
  const _Float16* nrow = X + (size_t)(b * NN + nbr) * ldin;
  _Float16* erow = &eT[w][r * CPAD2];
  if constexpr ((CIN % 8) == 0 && CPAD2 == 2 * CIN) {
    // packed-f16 vector path: v_pk_* + b128 LDS stores
    const v8h* nv = reinterpret_cast<const v8h*>(nrow);
    const v8h* xv = reinterpret_cast<const v8h*>(xrow);
    v8h* ev = reinterpret_cast<v8h*>(erow);
#pragma unroll
    for (int q = 0; q < CIN / 8; ++q) ev[q] = nv[q] - xv[q];
#pragma unroll
    for (int q = 0; q < CIN / 8; ++q) ev[CIN / 8 + q] = xv[q];
  } else {
#pragma unroll 4
    for (int cc = 0; cc < CPAD2; ++cc) {
      _Float16 v;
      if (cc < CIN)          v = nrow[cc] - xrow[cc];
      else if (cc < 2 * CIN) v = xrow[cc - CIN];
      else                   v = (_Float16)0.f;
      erow[cc] = v;
    }
  }
  __syncthreads();

  const int c  = lane & 15;
  const int mb = (lane < 16) ? 0 : 8;

  // stage 1: h = lrelu(W1 e + b1)  -> LDS
  for (int nt = 0; nt < COUT / 16; ++nt) {
    v8f acc0 = {}, acc1 = {};
    for (int kt = 0; kt < CPAD2 / 32; ++kt) {
      const v16h af0 = load_a(&eT[w][0 * CPAD2 + kt * 32], CPAD2);
      const v16h af1 = load_a(&eT[w][16 * CPAD2 + kt * 32], CPAD2);
      const v16h bf  = load_b(W1, CPAD2, nt * 16, kt * 32);
      acc0 = wmma(af0, bf, acc0);
      acc1 = wmma(af1, bf, acc1);
    }
    const int   col  = nt * 16 + c;
    const float bias = b1[col];
#pragma unroll
    for (int i = 0; i < 8; ++i) {
      hT[w][(mb + i) * COUT + col]      = (_Float16)lrelu(acc0[i] + bias);
      hT[w][(16 + mb + i) * COUT + col] = (_Float16)lrelu(acc1[i] + bias);
    }
  }
  __syncthreads();

  // stage 2: lrelu(W2 h + b2), then max over the 20 valid edges
  for (int nt = 0; nt < COUT / 16; ++nt) {
    v8f acc0 = {}, acc1 = {};
    for (int kt = 0; kt < COUT / 32; ++kt) {
      const v16h af0 = load_a(&hT[w][0 * COUT + kt * 32], COUT);
      const v16h af1 = load_a(&hT[w][16 * COUT + kt * 32], COUT);
      const v16h bf  = load_b(W2, COUT, nt * 16, kt * 32);
      acc0 = wmma(af0, bf, acc0);
      acc1 = wmma(af1, bf, acc1);
    }
    const int   col  = nt * 16 + c;
    const float bias = b2[col];
    float m = -3.4e38f;
#pragma unroll
    for (int i = 0; i < 8; ++i) {
      m = fmaxf(m, lrelu(acc0[i] + bias));          // rows 0..15 all valid
      const int row1 = 16 + mb + i;                 // rows 16..31
      if (row1 < KNBR) m = fmaxf(m, lrelu(acc1[i] + bias));
    }
    m = fmaxf(m, __shfl_xor(m, 16, 32));
    if (lane < 16)
      out[(size_t)(b * NN + n) * ldout + col] = (_Float16)m;
  }
}

// ---------------------------------------------------------------------------
// Generic WMMA GEMM + bias (+ optional LReLU).  One wave computes a 16-row x
// (NT*16)-col tile.  Weight tiles are TDM-staged into a per-wave LDS double
// buffer (tensor_load_to_lds / s_wait_tensorcnt); A2 (row-stride 0) supplies
// the broadcast global-feature half of the head-1 concat when k0 >= kSplit.
// ---------------------------------------------------------------------------
template <int KDIM, int COUT, bool LRELU, bool F32OUT>
__global__ __launch_bounds__(128) void gemm_kernel(
    const _Float16* __restrict__ A, int lda, const _Float16* __restrict__ A2,
    int kSplit, const _Float16* __restrict__ W, const float* __restrict__ bias,
    void* __restrict__ outv, int ldo, int coutStore) {
  constexpr int NT   = (COUT >= 64) ? 4 : (COUT / 16);
  constexpr int NBLK = COUT / (NT * 16);
  constexpr int KT   = KDIM / 32;
  const int wid     = (blockIdx.x << 2) + (threadIdx.x >> 5);
  const int wv      = threadIdx.x >> 5;
  const int rowTile = wid / NBLK;
  const int nb      = wid - rowTile * NBLK;
  const int rowbase = rowTile * 16;
  const int b       = rowbase / NN;
  const int lane    = threadIdx.x & 31;

  v8f acc[NT];
#pragma unroll
  for (int j = 0; j < NT; ++j) acc[j] = {};

#ifdef HAVE_TDM
  __shared__ alignas(16) _Float16 wstage[4][2][64 * 32];
  const _Float16* wsrc = W + (size_t)(nb * NT * 16) * KDIM;
  // prologue: stage k-tile 0 into buffer 0
  tdm_load_tile((unsigned)(uintptr_t)&wstage[wv][0][0], wsrc, 32, NT * 16, KDIM);
#endif

  for (int kt = 0; kt < KT; ++kt) {
    const int k0 = kt * 32;
    const _Float16* abase; int aldh;
    if (k0 < kSplit) { abase = A + (size_t)rowbase * lda + k0; aldh = lda; }
    else { abase = A2 + (size_t)b * (KDIM - kSplit) + (k0 - kSplit); aldh = 0; }
    const v16h af = load_a(abase, aldh);
#ifdef HAVE_TDM
    if (kt + 1 < KT) {
      tdm_load_tile((unsigned)(uintptr_t)&wstage[wv][(kt + 1) & 1][0],
                    wsrc + (kt + 1) * 32, 32, NT * 16, KDIM);
      __builtin_amdgcn_s_wait_tensorcnt(1);   // oldest (current) tile landed
    } else {
      __builtin_amdgcn_s_wait_tensorcnt(0);
    }
    const _Float16* wt = &wstage[wv][kt & 1][0];
#pragma unroll
    for (int j = 0; j < NT; ++j) {
      const v16h bf = load_b_lds(wt, j);
      acc[j] = wmma(af, bf, acc[j]);
    }
#else
    __builtin_prefetch(W + (size_t)(nb * NT * 16) * KDIM + k0 + 128, 0, 1);
#pragma unroll
    for (int j = 0; j < NT; ++j) {
      const v16h bf = load_b(W, KDIM, nb * (NT * 16) + j * 16, k0);
      acc[j] = wmma(af, bf, acc[j]);
    }
#endif
  }
  const int c  = lane & 15;
  const int mb = (lane < 16) ? 0 : 8;
#pragma unroll
  for (int j = 0; j < NT; ++j) {
    const int   col = nb * (NT * 16) + j * 16 + c;
    const float bs  = bias[col];
#pragma unroll
    for (int i = 0; i < 8; ++i) {
      float v = acc[j][i] + bs;
      if (LRELU) v = lrelu(v);
      const int row = rowbase + mb + i;
      if (F32OUT) {
        if (col < coutStore) ((float*)outv)[(size_t)row * ldo + col] = v;
      } else {
        ((_Float16*)outv)[(size_t)row * ldo + col] = (_Float16)v;
      }
    }
  }
}

// ---------------------------------------------------------------------------
// Host orchestration.
// ---------------------------------------------------------------------------
extern "C" void kernel_launch(void* const* d_in, const int* in_sizes, int n_in,
                              void* d_out, int out_size, void* d_ws, size_t ws_size,
                              hipStream_t stream) {
  (void)out_size; (void)ws_size;
  // Locate x vs params (setup_inputs insertion order assumed; disambiguate by
  // size in case the harness places x last).
  int xi, p0;
  if (n_in > 0 && in_sizes[0] == BB * 3 * NN) { xi = 0; p0 = 1; }
  else { xi = n_in - 1; p0 = 0; }
  const float* x = (const float*)d_in[xi];
  auto P = [&](int i) -> const float* { return (const float*)d_in[p0 + i]; };

  // params (insertion order): ec{1..4}: W1, bn1{g,b,m,v}, W2, bn2{g,b,m,v};
  // conv5: W, bn{g,b,m,v}; head: W1, bn1{..}, W2, bn2{..}, W3, b3.
  const int ecb[4] = {0, 10, 20, 30};
  const float* c5W = P(40);
  const float* hW1 = P(45);
  const float* hW2 = P(50);
  const float* hW3 = P(55);
  const float* hb3 = P(56);

  size_t off = 0;
  auto alloc = [&](size_t bytes) -> void* {
    void* p = (char*)d_ws + off;
    off += (bytes + 255) & ~(size_t)255;
    return p;
  };

  _Float16* xh  = (_Float16*)alloc((size_t)BB * NN * 32 * 2);
  float*    sqb = (float*)alloc((size_t)BB * NN * 4);
  int*      idx = (int*)alloc((size_t)BB * NN * KNBR * 4);
  _Float16* cat = (_Float16*)alloc((size_t)BB * NN * 512 * 2);
  _Float16* lf  = (_Float16*)alloc((size_t)BB * NN * 1024 * 2);
  _Float16* pmx = (_Float16*)alloc((size_t)BB * 16 * 1024 * 2);
  _Float16* gf  = (_Float16*)alloc((size_t)BB * 1024 * 2);
  _Float16* hh1 = (_Float16*)alloc((size_t)BB * NN * 512 * 2);
  _Float16* hh2 = (_Float16*)alloc((size_t)BB * NN * 256 * 2);

  _Float16* Wh[12]; float* Bs[12];
  auto pack = [&](int slot, const float* W, int cout, int cin, const float* g,
                  const float* bb, const float* m, const float* v,
                  const float* bplain, int coutpad, int kpad) {
    Wh[slot] = (_Float16*)alloc((size_t)coutpad * kpad * 2);
    Bs[slot] = (float*)alloc((size_t)coutpad * 4);
    const int total = coutpad * kpad;
    pack_weight<<<(total + 255) / 256, 256, 0, stream>>>(
        W, cout, cin, g, bb, m, v, bplain, Wh[slot], kpad, coutpad, Bs[slot]);
  };

  // EdgeConv weights (slot: 2*layer = W1, 2*layer+1 = W2)
  const int ecout[4] = {64, 64, 128, 256};
  const int ecin1[4] = {6, 128, 128, 256};    // 2*cin (real)
  const int eck1p[4] = {32, 128, 128, 256};   // padded K for W1
  for (int L = 0; L < 4; ++L) {
    const int base = ecb[L];
    pack(2 * L, P(base + 0), ecout[L], ecin1[L], P(base + 1), P(base + 2),
         P(base + 3), P(base + 4), nullptr, ecout[L], eck1p[L]);
    pack(2 * L + 1, P(base + 5), ecout[L], ecout[L], P(base + 6), P(base + 7),
         P(base + 8), P(base + 9), nullptr, ecout[L], ecout[L]);
  }
  pack(8, c5W, 1024, 512, P(41), P(42), P(43), P(44), nullptr, 1024, 512);
  pack(9, hW1, 512, 2048, P(46), P(47), P(48), P(49), nullptr, 512, 2048);
  pack(10, hW2, 256, 512, P(51), P(52), P(53), P(54), nullptr, 256, 512);
  pack(11, hW3, 33, 256, nullptr, nullptr, nullptr, nullptr, hb3, 48, 256);

  prep_x<<<(BB * NN + 255) / 256, 256, 0, stream>>>(x, xh, sqb);

  const dim3 knnG(BB * NN / 32 / 4), knnB(128);
  const dim3 edgG(BB * NN / 2),      edgB(64);
  const dim3 sqG((BB * NN + 255) / 256), sqB(256);

  // edge conv 1: (B,N,3) -> cat[:, 0:64]
  knn_kernel<32><<<knnG, knnB, 0, stream>>>(xh, 32, sqb, idx);
  edge_kernel<3, 32, 64><<<edgG, edgB, 0, stream>>>(
      xh, 32, idx, Wh[0], Bs[0], Wh[1], Bs[1], cat + 0, 512);

  // edge conv 2: cat[0:64] -> cat[64:128]
  compute_sq<<<sqG, sqB, 0, stream>>>(cat + 0, 512, 64, sqb);
  knn_kernel<64><<<knnG, knnB, 0, stream>>>(cat + 0, 512, sqb, idx);
  edge_kernel<64, 128, 64><<<edgG, edgB, 0, stream>>>(
      cat + 0, 512, idx, Wh[2], Bs[2], Wh[3], Bs[3], cat + 64, 512);

  // edge conv 3: cat[64:128] -> cat[128:256]
  compute_sq<<<sqG, sqB, 0, stream>>>(cat + 64, 512, 64, sqb);
  knn_kernel<64><<<knnG, knnB, 0, stream>>>(cat + 64, 512, sqb, idx);
  edge_kernel<64, 128, 128><<<edgG, edgB, 0, stream>>>(
      cat + 64, 512, idx, Wh[4], Bs[4], Wh[5], Bs[5], cat + 128, 512);

  // edge conv 4: cat[128:256] -> cat[256:512]
  compute_sq<<<sqG, sqB, 0, stream>>>(cat + 128, 512, 128, sqb);
  knn_kernel<128><<<knnG, knnB, 0, stream>>>(cat + 128, 512, sqb, idx);
  edge_kernel<128, 256, 256><<<edgG, edgB, 0, stream>>>(
      cat + 128, 512, idx, Wh[6], Bs[6], Wh[7], Bs[7], cat + 256, 512);

  // conv5: 512 -> 1024  (waves = 512 row-tiles * 16 nblocks)
  gemm_kernel<512, 1024, true, false><<<2048, 128, 0, stream>>>(
      cat, 512, nullptr, 512, Wh[8], Bs[8], lf, 1024, 1024);

  // global max pool over N (16-way partial + reduce)
  colmax_part<<<(BB * 16 * 1024 + 255) / 256, 256, 0, stream>>>(lf, pmx);
  colmax_fin<<<(BB * 1024 + 255) / 256, 256, 0, stream>>>(pmx, gf);

  // head 1: [lf ; gf] 2048 -> 512
  gemm_kernel<2048, 512, true, false><<<1024, 128, 0, stream>>>(
      lf, 1024, gf, 1024, Wh[9], Bs[9], hh1, 512, 512);
  // head 2: 512 -> 256
  gemm_kernel<512, 256, true, false><<<512, 128, 0, stream>>>(
      hh1, 512, nullptr, 512, Wh[10], Bs[10], hh2, 256, 256);
  // head 3: 256 -> 33 (padded 48), f32 logits
  gemm_kernel<256, 48, false, true><<<128, 128, 0, stream>>>(
      hh2, 256, nullptr, 256, Wh[11], Bs[11], d_out, 33, 33);
}
